// scg_conv_82643760709697
// MI455X (gfx1250) — matrix operand order repaired and checked by reading, not compiled, not run on hardware
//
#include <hip/hip_runtime.h>
#include <stdint.h>

#define NVOX 120000
#define COUT 64
#define KOFF 27

typedef __attribute__((ext_vector_type(16))) __bf16 v16bf;
typedef __attribute__((ext_vector_type(8)))  float  v8f;

union FragAB { v16bf v; uint4 q[2]; };

__device__ __forceinline__ unsigned short f2bf(float f) {
    unsigned int u = __float_as_uint(f);
    return (unsigned short)((u + 0x7FFFu + ((u >> 16) & 1u)) >> 16);
}

// ---------------------------------------------------------------------------
// Reorder a (K, CIN, 64) f32 weight tensor into fragment-native bf16 B tiles.
// Memory layout consumed by the conv kernel:
//   frag_id = (k*(CIN/32)+ks)*4 + nt ; element = frag_id*512 + lane*16 + i
// ISA B layout (16-bit, 32x16): lane<16 -> K {0..7,16..23}, lane>=16 -> K {8..15,24..31} ; N=lane&15
// ---------------------------------------------------------------------------
__global__ __launch_bounds__(256) void reorder_w_kern(const float* __restrict__ W,
                                                      unsigned short* __restrict__ Wfrag,
                                                      int CIN) {
    const int KS = CIN / 32;
    const int total = KOFF * KS * 4 * 512;
    const int id = blockIdx.x * 256 + threadIdx.x;
    if (id >= total) return;
    const int i    = id & 15;
    const int lane = (id >> 4) & 31;
    const int nt   = (id >> 9) & 3;
    const int t    = id >> 11;            // k*KS + ks
    const int ks   = t % KS;
    const int k    = t / KS;
    const int hi   = lane >> 4;
    const int c    = ks * 32 + (i >> 3) * 16 + hi * 8 + (i & 7);       // K row within B
    const int n    = nt * 16 + (lane & 15);       // N column
    Wfrag[id] = f2bf(W[((size_t)k * CIN + c) * COUT + n]);
}

// ---------------------------------------------------------------------------
// x_F (N,32) f32 -> bf16 padded with a zero sentinel row at index N.
// Also zeros the sentinel row of the Fs bf16 buffer used by conv2.
// ---------------------------------------------------------------------------
__global__ __launch_bounds__(256) void prep_x_kern(const float* __restrict__ xF,
                                                   unsigned short* __restrict__ xb,
                                                   unsigned short* __restrict__ Fsb) {
    const int id = blockIdx.x * 256 + threadIdx.x;
    const int totx = (NVOX + 1) * 32;
    if (id < totx) xb[id] = (id < NVOX * 32) ? f2bf(xF[id]) : (unsigned short)0;
    if (id < COUT) Fsb[(size_t)NVOX * COUT + id] = 0;
}

// ---------------------------------------------------------------------------
// Sparse 3^3 conv via WMMA. 64 threads = 2 waves per block; each wave owns a
// 32-row x 64-col output tile (two 16-row A tiles sharing every B fragment,
// halving weight traffic and doubling per-wave WMMA ILP). The nbr sentinel
// (== NVOX) hits the zero row, so the loop is branch-free and EXEC stays
// all-ones (WMMA requirement). 120000 = 64 * 1875 -> no partial tiles.
// MODE 0: dst = acc + bias (writes x1)
// MODE 1: dst += m_s * (acc + bias) (masked RMW into final output)
// ---------------------------------------------------------------------------
template <int CIN, int MODE>
__global__ __launch_bounds__(64) void sconv_wmma_kern(
    const unsigned short* __restrict__ Fin,    // (NVOX+1) x CIN bf16
    const unsigned short* __restrict__ Wfrag,  // fragment-native bf16 weights
    const int* __restrict__ nbr,               // NVOX x 27 kernel map
    const float* __restrict__ bias,            // COUT f32
    const float* __restrict__ msr,             // NVOX f32 (MODE 1 only)
    float* __restrict__ dst)                   // NVOX x COUT f32
{
    constexpr int KS = CIN / 32;
    const int lane = threadIdx.x & 31;
    const int wave = threadIdx.x >> 5;
    const int row0 = blockIdx.x * 64 + wave * 32;   // 32 rows per wave
    const int m    = lane & 15;
    const int hi   = lane >> 4;

    v8f acc[2][4] = {};
    const int* nrow0 = nbr + (size_t)(row0 + m) * KOFF;
    const int* nrow1 = nbr + (size_t)(row0 + 16 + m) * KOFF;

    for (int k = 0; k < KOFF; ++k) {
        const int idx0 = nrow0[k];                        // in [0, NVOX]
        const int idx1 = nrow1[k];
        const unsigned short* ar0 = Fin + (size_t)idx0 * CIN;
        const unsigned short* ar1 = Fin + (size_t)idx1 * CIN;
#pragma unroll
        for (int ks = 0; ks < KS; ++ks) {
            // ISA A layout (16-bit, 16x32): lane<16 holds K {0..7,16..23},
            // lane>=16 holds K {8..15,24..31} of row M=lane&15.
            FragAB a0, a1;
            const uint4* ap0 = (const uint4*)(ar0 + ks * 32);
            const uint4* ap1 = (const uint4*)(ar1 + ks * 32);
            a0.q[0] = ap0[hi];
            a0.q[1] = ap0[2 + hi];
            a1.q[0] = ap1[hi];
            a1.q[1] = ap1[2 + hi];
#pragma unroll
            for (int nt = 0; nt < 4; ++nt) {
                FragAB b;
                const uint4* bp =
                    (const uint4*)(Wfrag + ((size_t)(k * KS + ks) * 4 + nt) * 512 + lane * 16);
                b.q[0] = bp[0];
                b.q[1] = bp[1];
                acc[0][nt] = __builtin_amdgcn_wmma_f32_16x16x32_bf16(
                    false, a0.v, false, b.v, (short)0, acc[0][nt], false, false);
                acc[1][nt] = __builtin_amdgcn_wmma_f32_16x16x32_bf16(
                    false, a1.v, false, b.v, (short)0, acc[1][nt], false, false);
            }
        }
    }

    const int col0 = lane & 15;
#pragma unroll
    for (int t = 0; t < 2; ++t) {
        const int rbase = row0 + t * 16 + hi * 8;  // C layout: VGPR r -> M = r + hi*8
        if constexpr (MODE == 0) {
#pragma unroll
            for (int nt = 0; nt < 4; ++nt) {
                const int col = nt * 16 + col0;
                const float bv = bias[col];
#pragma unroll
                for (int r = 0; r < 8; ++r) {
                    dst[(size_t)(rbase + r) * COUT + col] = acc[t][nt][r] + bv;
                }
            }
        } else {
#pragma unroll
            for (int r = 0; r < 8; ++r) {
                const int row = rbase + r;
                const float ms = msr[row];
#pragma unroll
                for (int nt = 0; nt < 4; ++nt) {
                    const int col = nt * 16 + col0;
                    const size_t o = (size_t)row * COUT + col;
                    dst[o] += ms * (acc[t][nt][r] + bias[col]);
                }
            }
        }
    }
}

// ---------------------------------------------------------------------------
// Per-row routing: logits -> Gumbel softmax -> hard straight-through masks.
// Writes: Fs (bf16, conv2 input), partial out = m_w*Fw + x1, and m_s.
// ---------------------------------------------------------------------------
__global__ __launch_bounds__(256) void route_kern(const float* __restrict__ x1,
                                                  const float* __restrict__ gum,
                                                  const float* __restrict__ Wcls,
                                                  const float* __restrict__ bcls,
                                                  float* __restrict__ out,
                                                  unsigned short* __restrict__ Fsb,
                                                  float* __restrict__ msr) {
    const int r = blockIdx.x * 256 + threadIdx.x;
    if (r >= NVOX) return;
    const float* xr = x1 + (size_t)r * COUT;

    float l0 = bcls[0], l1 = bcls[1];
#pragma unroll 8
    for (int c = 0; c < COUT; ++c) {
        const float v = xr[c];
        l0 += v * Wcls[2 * c];
        l1 += v * Wcls[2 * c + 1];
    }
    const float a0 = (l0 + gum[2 * r]) * 10.0f;       // / TAU (=0.1)
    const float a1 = (l1 + gum[2 * r + 1]) * 10.0f;
    const float mx = fmaxf(a0, a1);
    const float e0 = __expf(a0 - mx), e1 = __expf(a1 - mx);
    const float inv = 1.0f / (e0 + e1);
    const float s0 = e0 * inv, s1 = e1 * inv;
    const float h0 = (a0 >= a1) ? 1.0f : 0.0f;        // argmax, first index wins ties
    const float h1 = 1.0f - h0;
    const float y0 = h0 + s0 - s0;                    // straight-through, reference FP order
    const float y1 = h1 + s1 - s1;
    msr[r] = h0;

    unsigned short* fs = Fsb + (size_t)r * COUT;
    float* op = out + (size_t)r * COUT;
#pragma unroll 8
    for (int c = 0; c < COUT; ++c) {
        const float v = xr[c];
        fs[c] = f2bf((v * y0) * h0);                  // strong-subset features
        op[c] = h1 * (v * y1) + v;                    // m_w*Fw + x1 (conv_s added later)
    }
}

extern "C" void kernel_launch(void* const* d_in, const int* in_sizes, int n_in,
                              void* d_out, int out_size, void* d_ws, size_t ws_size,
                              hipStream_t stream) {
    (void)in_sizes; (void)n_in; (void)out_size; (void)ws_size;
    const float* xF   = (const float*)d_in[0];
    const float* gum  = (const float*)d_in[1];
    const int*   nbr  = (const int*)d_in[2];
    const float* Wch  = (const float*)d_in[3];
    const float* bch  = (const float*)d_in[4];
    const float* Wcls = (const float*)d_in[5];
    const float* bcls = (const float*)d_in[6];
    const float* Wdw  = (const float*)d_in[7];
    const float* bdw  = (const float*)d_in[8];
    float* out = (float*)d_out;

    // Workspace carve-out (256B aligned), ~55 MB total.
    char* ws = (char*)d_ws;
    size_t off = 0;
    auto carve = [&](size_t bytes) {
        char* p = ws + off;
        off = (off + bytes + 255) & ~(size_t)255;
        return p;
    };
    float*          x1   = (float*)carve((size_t)NVOX * COUT * 4);
    unsigned short* xb   = (unsigned short*)carve((size_t)(NVOX + 1) * 32 * 2);
    unsigned short* Fsb  = (unsigned short*)carve((size_t)(NVOX + 1) * COUT * 2);
    unsigned short* wchf = (unsigned short*)carve((size_t)KOFF * 1 * 4 * 512 * 2);
    unsigned short* wdwf = (unsigned short*)carve((size_t)KOFF * 2 * 4 * 512 * 2);
    float*          msr  = (float*)carve((size_t)NVOX * 4);

    // 1) weight swizzle + input bf16 conversion
    reorder_w_kern<<<(KOFF * 1 * 4 * 512 + 255) / 256, 256, 0, stream>>>(Wch, wchf, 32);
    reorder_w_kern<<<(KOFF * 2 * 4 * 512 + 255) / 256, 256, 0, stream>>>(Wdw, wdwf, 64);
    prep_x_kern<<<(((NVOX + 1) * 32) + 255) / 256, 256, 0, stream>>>(xF, xb, Fsb);

    // 2) conv1 (Channel_code): x1 = sconv(x_F, W_ch) + b_ch
    sconv_wmma_kern<32, 0><<<NVOX / 64, 64, 0, stream>>>(xb, wchf, nbr, bch, nullptr, x1);

    // 3) routing: masks, Fs, partial output
    route_kern<<<(NVOX + 255) / 256, 256, 0, stream>>>(x1, gum, Wcls, bcls, out, Fsb, msr);

    // 4) conv2 on strong subset, masked accumulate into final output
    sconv_wmma_kern<64, 1><<<NVOX / 64, 64, 0, stream>>>(Fsb, wdwf, nbr, bdw, msr, out);
}
